// MultiHeadAttention_33663953666992
// MI455X (gfx1250) — compile-verified, matrix operand order checked by reading, not run on hardware
//
#include <hip/hip_runtime.h>
#include <math.h>

// ---------------------------------------------------------------------------
// Types for CDNA5 WMMA (wave32): v_wmma_f32_16x16x32_bf16
// ---------------------------------------------------------------------------
typedef __bf16 bf16_t;
typedef __attribute__((ext_vector_type(16))) __bf16 v16bf;
typedef __attribute__((ext_vector_type(8)))  __bf16 v8bf;
typedef __attribute__((ext_vector_type(8)))  float  v8f;

static __device__ __forceinline__ v8f wmma_bf16(v16bf a, v16bf b, v8f c) {
    // 8 args: (neg_a, A, neg_b, B, c_mod, C, reuse_a, reuse_b)
    return __builtin_amdgcn_wmma_f32_16x16x32_bf16(
        false, a, false, b, (short)0, c, false, false);
}

// Load a 16x32 (rows x K) bf16 fragment from row-major storage, ld = leading dim.
// ISA 7.12.2 (16-bit A 16x32; B mirrored with N in place of M):
//   lanes 0-15 : row = base+lane,    K chunks {k0..k0+7, k0+16..k0+23}
//   lanes 16-31: row = base+lane-16, K chunks {k0+8..k0+15, k0+24..k0+31}
// Two global_load_b128 per lane.
static __device__ __forceinline__ v16bf load_frag_rowK(
    const bf16_t* __restrict__ base, int ld, int row0, int k0, int lane) {
    const int r    = lane & 15;
    const int kofs = (lane >> 4) ? 8 : 0;
    const bf16_t* p0 = base + (size_t)(row0 + r) * ld + (k0 + kofs);
    v8bf a = *(const v8bf*)p0;
    v8bf b = *(const v8bf*)(p0 + 16);
    v16bf out;
#pragma unroll
    for (int i = 0; i < 8; ++i) { out[i] = a[i]; out[i + 8] = b[i]; }
    return out;
}

// ---------------------------------------------------------------------------
// Packing / conversion kernels
// ---------------------------------------------------------------------------
__global__ void mha_cvt_bf16(const float* __restrict__ in, bf16_t* __restrict__ out, int n) {
    int i = blockIdx.x * blockDim.x + threadIdx.x;
    if (i < n) out[i] = (bf16_t)in[i];
}

// W[h][c][d] (f32) -> Wt[h][d][c] (bf16)  => flat [N=h*D+d][K=c] row-major
__global__ void mha_pack_w(const float* __restrict__ W, bf16_t* __restrict__ Wt,
                           int total, int C, int D) {
    int i = blockIdx.x * blockDim.x + threadIdx.x;
    if (i >= total) return;
    int d = i % D;
    int c = (i / D) % C;
    int h = i / (D * C);
    Wt[((size_t)h * D + d) * C + c] = (bf16_t)W[i];
}

// Wp[k][n] (f32) -> WpT[n][k] (bf16)
__global__ void mha_pack_wp(const float* __restrict__ Wp, bf16_t* __restrict__ WpT, int C) {
    int i = blockIdx.x * blockDim.x + threadIdx.x;
    if (i >= C * C) return;
    int n = i % C;
    int k = i / C;
    WpT[(size_t)n * C + k] = (bf16_t)Wp[i];
}

// ---------------------------------------------------------------------------
// WMMA GEMM:  out[m,n] = sum_k A[m,k]*B[n,k] + bias[n]
//   A: [M,K] bf16 row-major, B: [N,K] bf16 row-major.
// Block: 256 threads = 8 waves (2M x 4N) -> 64(M) x 256(N) tile; wave 32x64.
// Store modes: 0 = bf16 into [B,H,T,D] (k,q), 1 = bf16 into [B,H,D,T] (v^T),
//              2 = f32 row-major [M,N] (final output).
// ---------------------------------------------------------------------------
template <int MODE>
__global__ void mha_gemm_wmma(const bf16_t* __restrict__ A,
                              const bf16_t* __restrict__ Bm,
                              const float* __restrict__ bias,
                              void* __restrict__ out,
                              int M, int N, int K, int T, int H) {
    const int lane = threadIdx.x & 31;
    const int wave = threadIdx.x >> 5;
    const int wm = wave & 1;   // 2 waves along M
    const int wn = wave >> 1;  // 4 waves along N
    const int m0 = blockIdx.x * 64 + wm * 32;
    const int n0 = blockIdx.y * 256 + wn * 64;

    v8f acc[2][4] = {};
    for (int k0 = 0; k0 < K; k0 += 32) {
        v16bf a0 = load_frag_rowK(A, K, m0,      k0, lane);
        v16bf a1 = load_frag_rowK(A, K, m0 + 16, k0, lane);
#pragma unroll
        for (int j = 0; j < 4; ++j) {
            v16bf b = load_frag_rowK(Bm, K, n0 + 16 * j, k0, lane);
            acc[0][j] = wmma_bf16(a0, b, acc[0][j]);
            acc[1][j] = wmma_bf16(a1, b, acc[1][j]);
        }
    }

    // C/D layout: lane L, vgpr r -> row m0+16i + r + 8*(L>=16), col n0+16j+(L&15)
    const int half = lane >> 4;
    const int nc = lane & 15;
#pragma unroll
    for (int i = 0; i < 2; ++i) {
#pragma unroll
        for (int j = 0; j < 4; ++j) {
#pragma unroll
            for (int r = 0; r < 8; ++r) {
                const int m = m0 + 16 * i + r + 8 * half;
                const int n = n0 + 16 * j + nc;
                const float val = acc[i][j][r] + bias[n];
                if (MODE == 2) {
                    ((float*)out)[(size_t)m * N + n] = val;
                } else {
                    const int b = m / T, t = m % T;
                    const int h = n >> 6, d = n & 63;
                    size_t idx;
                    if (MODE == 0) idx = (((size_t)b * H + h) * T + t) * 64 + d;        // [B,H,T,D]
                    else           idx = (((size_t)b * H + h) * 64 + d) * (size_t)T + t; // [B,H,D,T]
                    ((bf16_t*)out)[idx] = (bf16_t)val;
                }
            }
        }
    }
}

// ---------------------------------------------------------------------------
// Flash attention, fully transposed formulation.
//   S^T[s,t] = q[s].k[t]  (A = q rows, B = k cols)  -> lane owns column t
//   O^T[d,t] = sum_s v^T[d,s] * P^T[s,t]            (A = v^T rows, B = P^T)
// Key property: the S^T C-fragment element (tile m, vgpr r, half) is exactly
// the B-fragment slot (i = 8m+r, K-chunk ofs 8*half) for the O^T WMMA, so P
// stays in registers (no LDS). Softmax stats are per-lane (t = lane&15):
// reduction = 15 VALU ops + ONE shfl_xor(16) per {max,sum}.
// One wave owns 16 t-columns; block = 4 waves = 64 t.
// ---------------------------------------------------------------------------
__global__ void mha_flash_attn(const bf16_t* __restrict__ Kp,
                               const bf16_t* __restrict__ Qp,
                               const bf16_t* __restrict__ VTp,
                               bf16_t* __restrict__ Oh,
                               int H, int T, float scale) {
    const int lane = threadIdx.x & 31;
    const int wave = threadIdx.x >> 5;
    const int bh = blockIdx.y;
    const int b = bh / H, h = bh % H;
    const int t0 = blockIdx.x * 64 + wave * 16;

    const bf16_t* kbase = Kp  + (size_t)bh * T * 64;
    const bf16_t* qbase = Qp  + (size_t)bh * T * 64;
    const bf16_t* vbase = VTp + (size_t)bh * 64 * T;

    // k tile as B operand (N = t columns, K = d), resident for whole loop
    const v16bf kb0 = load_frag_rowK(kbase, 64, t0, 0,  lane);
    const v16bf kb1 = load_frag_rowK(kbase, 64, t0, 32, lane);

    float mstat = -INFINITY, lstat = 0.f;
    v8f o[4] = {};   // O^T: 4 d-tiles x (8 vgprs), col t = lane&15

    for (int s0 = 0; s0 < T; s0 += 32) {
        // ---- S^T: two 16-row s-tiles, D=64 in two WMMA K-steps each ----
        v8f st[2];
#pragma unroll
        for (int m = 0; m < 2; ++m) {
            v8f accs = {};
            v16bf qa0 = load_frag_rowK(qbase, 64, s0 + 16 * m, 0,  lane);
            accs = wmma_bf16(qa0, kb0, accs);
            v16bf qa1 = load_frag_rowK(qbase, 64, s0 + 16 * m, 32, lane);
            accs = wmma_bf16(qa1, kb1, accs);
            st[m] = accs;
        }
        // ---- per-lane online softmax over this lane's 16 s-values ----
        float x[16];
#pragma unroll
        for (int m = 0; m < 2; ++m)
#pragma unroll
            for (int r = 0; r < 8; ++r)
                x[8 * m + r] = st[m][r] * scale;

        float mx = x[0];
#pragma unroll
        for (int i = 1; i < 16; ++i) mx = fmaxf(mx, x[i]);
        mx = fmaxf(mx, __shfl_xor(mx, 16, 32));   // combine the two halves
        const float mnew = fmaxf(mstat, mx);
        const float alpha = __expf(mstat - mnew);

        v16bf pb;           // P^T B-fragment, built in-register
        float psum = 0.f;
#pragma unroll
        for (int i = 0; i < 16; ++i) {
            const float p = __expf(x[i] - mnew);
            psum += p;
            pb[i] = (bf16_t)p;
        }
        psum += __shfl_xor(psum, 16, 32);
        lstat = lstat * alpha + psum;
        mstat = mnew;

        // rescale O^T (alpha is uniform per lane / per t-column)
#pragma unroll
        for (int j = 0; j < 4; ++j)
#pragma unroll
            for (int r = 0; r < 8; ++r) o[j][r] *= alpha;

        // ---- O^T += V^T (d x 32s) x P^T (32s x 16t) ----
#pragma unroll
        for (int j = 0; j < 4; ++j) {
            v16bf va = load_frag_rowK(vbase, T, 16 * j, s0, lane);
            o[j] = wmma_bf16(va, pb, o[j]);
        }
    }

    // ---- normalize, store concat-head output [B,T,H*64] ----
    // lane owns column t = t0+nc; d = 16j + r + 8*half -> 8 consecutive d per j
    const int half = lane >> 4;
    const int nc = lane & 15;
    const float inv = 1.0f / lstat;
    const size_t rowbase = ((size_t)b * T + (t0 + nc)) * (size_t)(H * 64) + (size_t)h * 64;
#pragma unroll
    for (int j = 0; j < 4; ++j) {
        v8bf ov;
#pragma unroll
        for (int r = 0; r < 8; ++r) ov[r] = (bf16_t)(o[j][r] * inv);
        *(v8bf*)(Oh + rowbase + 16 * j + 8 * half) = ov;
    }
}

// ---------------------------------------------------------------------------
// Host-side launch
// ---------------------------------------------------------------------------
extern "C" void kernel_launch(void* const* d_in, const int* in_sizes, int n_in,
                              void* d_out, int out_size, void* d_ws, size_t ws_size,
                              hipStream_t stream) {
    constexpr int Bb = 2, T = 2048, Cc = 1024, H = 16, D = 64;
    constexpr int M = Bb * T;          // 4096
    const float scale = 1.0f / 32.0f;  // C^-0.5 = 1024^-0.5

    const float* x  = (const float*)d_in[0];
    const float* Wk = (const float*)d_in[1];
    const float* bk = (const float*)d_in[2];
    const float* Wq = (const float*)d_in[3];
    const float* bq = (const float*)d_in[4];
    const float* Wv = (const float*)d_in[5];
    const float* bv = (const float*)d_in[6];
    const float* Wp = (const float*)d_in[7];
    const float* bp = (const float*)d_in[8];

    char* ws = (char*)d_ws;
    size_t off = 0;
    bf16_t* xb   = (bf16_t*)(ws + off); off += (size_t)M * Cc * sizeof(bf16_t);         // 8 MB
    bf16_t* wkT  = (bf16_t*)(ws + off); off += (size_t)H * D * Cc * sizeof(bf16_t);     // 2 MB
    bf16_t* wqT  = (bf16_t*)(ws + off); off += (size_t)H * D * Cc * sizeof(bf16_t);
    bf16_t* wvT  = (bf16_t*)(ws + off); off += (size_t)H * D * Cc * sizeof(bf16_t);
    bf16_t* wpT  = (bf16_t*)(ws + off); off += (size_t)Cc * Cc * sizeof(bf16_t);        // 2 MB
    bf16_t* kmat = (bf16_t*)(ws + off); off += (size_t)Bb * H * T * D * sizeof(bf16_t); // 8 MB
    bf16_t* qmat = (bf16_t*)(ws + off); off += (size_t)Bb * H * T * D * sizeof(bf16_t);
    bf16_t* vTm  = (bf16_t*)(ws + off); off += (size_t)Bb * H * T * D * sizeof(bf16_t);
    bf16_t* oh   = (bf16_t*)(ws + off); off += (size_t)M * Cc * sizeof(bf16_t);         // 8 MB

    // 1) conversions / packing
    {
        int n = M * Cc;
        mha_cvt_bf16<<<(n + 255) / 256, 256, 0, stream>>>(x, xb, n);
        int nw = H * Cc * D;
        mha_pack_w<<<(nw + 255) / 256, 256, 0, stream>>>(Wk, wkT, nw, Cc, D);
        mha_pack_w<<<(nw + 255) / 256, 256, 0, stream>>>(Wq, wqT, nw, Cc, D);
        mha_pack_w<<<(nw + 255) / 256, 256, 0, stream>>>(Wv, wvT, nw, Cc, D);
        int np = Cc * Cc;
        mha_pack_wp<<<(np + 255) / 256, 256, 0, stream>>>(Wp, wpT, Cc);
    }

    // 2) QKV projections: M=4096, N=1024 (h*64+d), K=1024
    {
        dim3 grid(M / 64, (H * D) / 256);
        mha_gemm_wmma<0><<<grid, 256, 0, stream>>>(xb, wkT, bk, (void*)kmat, M, H * D, Cc, T, H);
        mha_gemm_wmma<0><<<grid, 256, 0, stream>>>(xb, wqT, bq, (void*)qmat, M, H * D, Cc, T, H);
        mha_gemm_wmma<1><<<grid, 256, 0, stream>>>(xb, wvT, bv, (void*)vTm,  M, H * D, Cc, T, H);
    }

    // 3) flash attention: grid (T/64 tiles of t, B*H), 4 waves/block
    {
        dim3 grid(T / 64, Bb * H);
        mha_flash_attn<<<grid, 128, 0, stream>>>(kmat, qmat, vTm, oh, H, T, scale);
    }

    // 4) output projection: [4096,1024] x [1024,1024]^T + bp -> f32 d_out
    {
        dim3 grid(M / 64, Cc / 256);
        mha_gemm_wmma<2><<<grid, 256, 0, stream>>>(oh, wpT, bp, d_out, M, Cc, Cc, T, H);
    }

    (void)in_sizes; (void)n_in; (void)out_size; (void)ws_size;
}
